// ElectronicSpatialExtentDecoder_27625229648413
// MI455X (gfx1250) — matrix operand order repaired and checked by reading, not compiled, not run on hardware
//
#include <hip/hip_runtime.h>
#include <hip/hip_bf16.h>

// Problem constants (from reference): N=100000, B=128, F=512, H=256
#define FDIM 512
#define HDIM 256
#define NT   (HDIM / 16)   // 16 n-tiles of 16 columns
#define KC   (FDIM / 32)   // 16 k-chunks of 32 (bf16 WMMA K)

typedef __attribute__((ext_vector_type(16))) __bf16 v16bf;
typedef __attribute__((ext_vector_type(8)))  float  v8f;
typedef __attribute__((ext_vector_type(4)))  float  v4f;

// ---------------------------------------------------------------------------
// Prep: W1 [F][H] fp32 (row-major) -> W1t [H][F] bf16. 256 KB, L2-resident.
// Consecutive threads read consecutive W1 columns (coalesced reads).
// ---------------------------------------------------------------------------
__global__ void w1_transpose_bf16(const float* __restrict__ W1,
                                  __bf16* __restrict__ W1t) {
    const int k = blockIdx.x;    // 0..F-1
    const int h = threadIdx.x;   // 0..H-1
    W1t[(size_t)h * FDIM + k] = (__bf16)W1[(size_t)k * HDIM + h];
}

__global__ void zero_out_kernel(float* __restrict__ out, int n) {
    const int i = blockIdx.x * blockDim.x + threadIdx.x;
    if (i < n) out[i] = 0.0f;
}

// ---------------------------------------------------------------------------
// Main fused kernel: 256 threads (8 wave32s), 128 node-rows per block.
// Each wave: 16 rows x 256 cols of h via v_wmma_f32_16x16x32_bf16,
// then fused silu -> dot(W2) -> * r2 -> atomic scatter-add into out[batch].
// ---------------------------------------------------------------------------
__global__ __launch_bounds__(256)
void esed_main(const float* __restrict__ pos,
               const float* __restrict__ mc,
               const float* __restrict__ scaler,
               const int*   __restrict__ batch,
               const __bf16* __restrict__ W1t,
               const float* __restrict__ b1,
               const float* __restrict__ W2,
               const float* __restrict__ b2p,
               float* __restrict__ out,
               int N)
{
    const int tid  = threadIdx.x;
    const int wave = tid >> 5;
    const int lane = tid & 31;
    const int half = lane >> 4;   // which 16-lane half of the wave
    const int l16  = lane & 15;

    const int rowBase = blockIdx.x * 128 + wave * 16;

    // --- A operand addressing (16-bit A, 16x32 tile; kernel's own K order, not the ISA table order):
    // lane (l16) holds row M=l16; contiguous K range starting at (lane/16)*16 (matches B).
    int arow = rowBase + l16;
    if (arow >= N) arow = N - 1;                 // clamp loads; writes are guarded
    const float* aptr = scaler + (size_t)arow * FDIM + half * 16;

    // --- B operand addressing (bf16, 32x16 tile): lane holds column n=l16,
    // contiguous K range starting at (lane/16)*16 -> one 32-byte load per tile.
    const __bf16* bbase = W1t + (size_t)l16 * FDIM + half * 16;

    v8f acc[NT] = {};   // 16 fp32 accumulator tiles = 128 VGPRs/lane

    for (int kc = 0; kc < KC; ++kc) {
        // Prefetch next K-chunk of this lane's scaler row (global_prefetch_b8)
        if (kc + 1 < KC) __builtin_prefetch(aptr + (kc + 1) * 32, 0, 0);

        // Load fp32 A fragment (streams HBM exactly once) and convert to bf16
        const float* ap = aptr + kc * 32;
        v4f f0 = *(const v4f*)(ap + 0);
        v4f f1 = *(const v4f*)(ap + 4);
        v4f f2 = *(const v4f*)(ap + 8);
        v4f f3 = *(const v4f*)(ap + 12);
        v16bf afrag;
#pragma unroll
        for (int i = 0; i < 4; ++i) {
            afrag[i]      = (__bf16)f0[i];
            afrag[4 + i]  = (__bf16)f1[i];
            afrag[8 + i]  = (__bf16)f2[i];
            afrag[12 + i] = (__bf16)f3[i];
        }

        // 16 independent WMMAs per K-chunk (B tiles hit L0/L2: W1t is 256 KB)
#pragma unroll
        for (int t = 0; t < NT; ++t) {
            v16bf bfrag = *(const v16bf*)(bbase + (size_t)t * 16 * FDIM + kc * 32);
            acc[t] = __builtin_amdgcn_wmma_f32_16x16x32_bf16(
                false, afrag, false, bfrag, (short)0, acc[t], false, false);
        }
    }

    // --- Fused epilogue: q[m] = sum_n silu(h[m][n]+b1[n]) * W2[n] + b2
    // C/D layout: VGPR r -> row M = 8*half + r ; column N = l16 (per tile t: n = t*16+l16)
    const float b2v = b2p[0];
    float p[8];
#pragma unroll
    for (int r = 0; r < 8; ++r) p[r] = 0.0f;

#pragma unroll
    for (int t = 0; t < NT; ++t) {
        const int n   = t * 16 + l16;
        const float b1n = b1[n];
        const float w2n = W2[n];
#pragma unroll
        for (int r = 0; r < 8; ++r) {
            float x = acc[t][r] + b1n;
            float s = x / (1.0f + __expf(-x));   // silu
            p[r] += s * w2n;
        }
    }

    // Reduce across the 16 lanes of each half (xor over lane bits 0..3 stays in-half)
#pragma unroll
    for (int r = 0; r < 8; ++r) {
        float v = p[r];
        v += __shfl_xor(v, 1, 32);
        v += __shfl_xor(v, 2, 32);
        v += __shfl_xor(v, 4, 32);
        v += __shfl_xor(v, 8, 32);
        p[r] = v;
    }

    // One lane per half finalizes its 8 rows: r2 * q -> atomic segment-sum
    if (l16 == 0) {
#pragma unroll
        for (int r = 0; r < 8; ++r) {
            const int row = rowBase + half * 8 + r;
            if (row < N) {
                const int b = batch[row];
                const float dx = pos[3 * row + 0] - mc[3 * b + 0];
                const float dy = pos[3 * row + 1] - mc[3 * b + 1];
                const float dz = pos[3 * row + 2] - mc[3 * b + 2];
                const float r2 = dx * dx + dy * dy + dz * dz;
                atomicAdd(&out[b], (p[r] + b2v) * r2);
            }
        }
    }
}

// ---------------------------------------------------------------------------
// Launch: inputs in setup_inputs() order:
// 0 pos[N,3] 1 mass_center[B,3] 2 scaler[N,F] 3 vector[N,3] (unused)
// 4 batch_index[N] 5 W1[F,H] 6 b1[H] 7 W2[H,1] 8 b2[1]
// ---------------------------------------------------------------------------
extern "C" void kernel_launch(void* const* d_in, const int* in_sizes, int n_in,
                              void* d_out, int out_size, void* d_ws, size_t ws_size,
                              hipStream_t stream) {
    const float* pos    = (const float*)d_in[0];
    const float* mc     = (const float*)d_in[1];
    const float* scaler = (const float*)d_in[2];
    const int*   batch  = (const int*)  d_in[4];
    const float* W1     = (const float*)d_in[5];
    const float* b1     = (const float*)d_in[6];
    const float* W2     = (const float*)d_in[7];
    const float* b2     = (const float*)d_in[8];
    float* out = (float*)d_out;

    __bf16* W1t = (__bf16*)d_ws;     // 256 KB of scratch
    const int N = in_sizes[4];       // node count

    hipLaunchKernelGGL(w1_transpose_bf16, dim3(FDIM), dim3(HDIM), 0, stream, W1, W1t);
    hipLaunchKernelGGL(zero_out_kernel, dim3(1), dim3(256), 0, stream, out, out_size);

    const int nblk = (N + 127) / 128;
    hipLaunchKernelGGL(esed_main, dim3(nblk), dim3(256), 0, stream,
                       pos, mc, scaler, batch, W1t, b1, W2, b2, out, N);
}
